// ProteinEncoder_49306224558162
// MI455X (gfx1250) — matrix operand order repaired and checked by reading, not compiled, not run on hardware
//
#include <hip/hip_runtime.h>
#include <math.h>

// ---------------------------------------------------------------------------
// Problem sizes (fixed by the reference)
// ---------------------------------------------------------------------------
#define N_NODES 1024
#define LDIM    128      // ligand dim / head dim
#define PDIM    1280     // protein embedding dim
#define NPROT   16
#define PLEN    512

// ---------------------------------------------------------------------------
// CDNA5 WMMA plumbing (wave32, v_wmma_f32_16x16x32_bf16)
// ---------------------------------------------------------------------------
typedef __attribute__((ext_vector_type(16))) __bf16          v16bf;
typedef __attribute__((ext_vector_type(8)))  float           v8f;
typedef __attribute__((ext_vector_type(16))) unsigned short  v16us;
typedef __attribute__((ext_vector_type(8)))  unsigned short  v8us;

static __device__ __forceinline__ unsigned short f2b(float f) {
    // fp32 -> bf16, round-to-nearest-even
    unsigned int u = __builtin_bit_cast(unsigned int, f);
    unsigned int r = u + 0x7FFFu + ((u >> 16) & 1u);
    return (unsigned short)(r >> 16);
}

static __device__ __forceinline__ v8f zero8() {
    v8f z;
#pragma unroll
    for (int i = 0; i < 8; ++i) z[i] = 0.0f;
    return z;
}

static __device__ __forceinline__ v16us pack16(v8us lo, v8us hi) {
    v16us t;
#pragma unroll
    for (int j = 0; j < 8; ++j) { t[j] = lo[j]; t[j + 8] = hi[j]; }
    return t;
}

static __device__ __forceinline__ v8f wmma_bf16(v16us a, v16us b, v8f c) {
    return __builtin_amdgcn_wmma_f32_16x16x32_bf16(
        false, __builtin_bit_cast(v16bf, a),
        false, __builtin_bit_cast(v16bf, b),
        (short)0, c, false, false);
}

// Fragment layout (CDNA5 ISA 7.12.2, wave32):
//   A (16x32 bf16): lane l -> row m = l&15, half = l>>4.
//       elem j -> k = 16*half + j (contiguous order used here, not the ISA table order)
//   B (32x16 bf16): lane l -> col n = l&15, half = l>>4; elem j -> k = 16*half + j (same contiguous order as A; this code's choice)
//   C/D (16x16 f32): lane l -> col n = l&15; vgpr r -> row m = r + 8*(l>>4)

// ---------------------------------------------------------------------------
// Tensor Data Mover (TDM): 2D tile DMA global(f32) -> LDS  (ISA ch.8 D#)
// ---------------------------------------------------------------------------
#if defined(__has_builtin)
#if __has_builtin(__builtin_amdgcn_tensor_load_to_lds)
#define HAVE_TDM 1
#endif
#endif

#ifdef HAVE_TDM
typedef __attribute__((ext_vector_type(4))) unsigned int v4u32;
typedef __attribute__((ext_vector_type(8))) int          v8i32;
typedef __attribute__((ext_vector_type(4))) int          v4i32;

// Load a tile of (tile_y rows) x (tile_x f32) from a row-major f32 tensor
// (row stride = stride_x elements) into LDS at byte offset lds_off,
// packed densely row-major.
static __device__ __forceinline__ void tdm_load_2d(unsigned lds_off,
                                                   const void* gptr,
                                                   unsigned tile_x, unsigned tile_y,
                                                   unsigned tensor_x, unsigned tensor_y,
                                                   unsigned stride_x) {
    unsigned long long ga = (unsigned long long)(size_t)gptr;
    v4u32 g0;
    g0[0] = 1u;                                   // count=1, no gather/restore
    g0[1] = lds_off;                              // lds_addr (bytes)
    g0[2] = (unsigned)ga;                         // global_addr[31:0]
    g0[3] = (unsigned)((ga >> 32) & 0x1FFFFFFull) // global_addr[56:32]
          | (2u << 30);                           // type = 2 ("image")
    v8i32 g1;
    g1[0] = (int)(2u << 16);                      // workgroup_mask=0, data_size=2 (4B)
    g1[1] = (int)((tensor_x & 0xFFFFu) << 16);    // tensor_dim0[15:0] @ bits 63:48
    g1[2] = (int)((tensor_x >> 16) & 0xFFFFu)     // tensor_dim0[31:16]
          | (int)((tensor_y & 0xFFFFu) << 16);    // tensor_dim1[15:0]
    g1[3] = (int)((tensor_y >> 16) & 0xFFFFu)     // tensor_dim1[31:16]
          | (int)((tile_x & 0xFFFFu) << 16);      // tile_dim0
    g1[4] = (int)(tile_y & 0xFFFFu);              // tile_dim1 (tile_dim2=0)
    g1[5] = (int)stride_x;                        // tensor_dim0_stride[31:0]
    g1[6] = 0;                                    // stride[47:32]=0, dim1_stride lo=0
    g1[7] = 0;
    v4i32 z4; z4[0] = 0; z4[1] = 0; z4[2] = 0; z4[3] = 0;
#if defined(__clang_major__) && (__clang_major__ >= 23)
    v8i32 z8;
#pragma unroll
    for (int i = 0; i < 8; ++i) z8[i] = 0;
    __builtin_amdgcn_tensor_load_to_lds(g0, g1, z4, z4, z8, 0);
#else
    __builtin_amdgcn_tensor_load_to_lds(g0, g1, z4, z4, 0);
#endif
}

static __device__ __forceinline__ void tdm_wait0() {
#if __has_builtin(__builtin_amdgcn_s_wait_tensorcnt)
    __builtin_amdgcn_s_wait_tensorcnt(0);
#else
    asm volatile("s_wait_tensorcnt 0x0" ::: "memory");
#endif
}
static __device__ __forceinline__ void tdm_wait3() {
#if __has_builtin(__builtin_amdgcn_s_wait_tensorcnt)
    __builtin_amdgcn_s_wait_tensorcnt(3);
#else
    asm volatile("s_wait_tensorcnt 0x3" ::: "memory");
#endif
}
#endif // HAVE_TDM

// ---------------------------------------------------------------------------
// Kernel 1: segment offsets of the sorted batch ids (lower bounds)
// ---------------------------------------------------------------------------
__global__ void seg_kernel(const int* __restrict__ batch, int* __restrict__ seg) {
    int t = threadIdx.x;
    if (t <= NPROT) {
        int c = 0;
        for (int i = 0; i < N_NODES; ++i) c += (batch[i] < t) ? 1 : 0;
        seg[t] = c;   // seg[0]=0 .. seg[16]=N_NODES
    }
}

// ---------------------------------------------------------------------------
// Kernel 2: Q = h_lig @ Wq + bq  -> bf16 [N_NODES][LDIM]
// grid.x = N_NODES/16 ; block = 256 (8 waves, wave w owns d-tile w)
// ---------------------------------------------------------------------------
__global__ void q_kernel(const float* __restrict__ h_lig,
                         const float* __restrict__ Wq,
                         const float* __restrict__ bq,
                         unsigned short* __restrict__ Qbf) {
    __shared__ unsigned short sA[16 * 32];    // [m][k]  (A-tile, bf16)
    __shared__ unsigned short sBT[128 * 32];  // [d][k]  (W-tile transposed, bf16)

    const int tid   = threadIdx.x;
    const int dt    = tid >> 5;        // wave id -> d-tile 0..7
    const int lane  = tid & 31;
    const int mlane = lane & 15;
    const int half  = lane >> 4;
    const int n0    = blockIdx.x * 16; // node tile base

    v8f acc = zero8();

    for (int e0 = 0; e0 < LDIM; e0 += 32) {
#pragma unroll
        for (int i = 0; i < 2; ++i) {
            int idx = tid + 256 * i;               // 512 elems
            int m = idx >> 5, k = idx & 31;
            sA[idx] = f2b(h_lig[(size_t)(n0 + m) * LDIM + e0 + k]);
        }
#pragma unroll
        for (int i = 0; i < 16; ++i) {
            int idx = tid + 256 * i;               // 4096 elems
            int k = idx >> 7, d = idx & 127;
            sBT[d * 32 + k] = f2b(Wq[(size_t)(e0 + k) * LDIM + d]);
        }
        __syncthreads();

        v8us  alo = *(const v8us*)&sA[mlane * 32 + 16 * half];
        v8us  ahi = *(const v8us*)&sA[mlane * 32 + 16 * half + 8];
        v16us a   = pack16(alo, ahi);
        v16us bfr = *(const v16us*)&sBT[(dt * 16 + mlane) * 32 + 16 * half];

        acc = wmma_bf16(a, bfr, acc);
        __syncthreads();
    }

    const int d    = dt * 16 + mlane;
    const float bias = bq[d];
#pragma unroll
    for (int r = 0; r < 8; ++r) {
        int row = n0 + r + 8 * half;
        Qbf[(size_t)row * LDIM + d] = f2b(acc[r] + bias);
    }
}

// ---------------------------------------------------------------------------
// Kernel 3: K,V = protein_embs @ {Wk,Wv} + {bk,bv}   (protein read ONCE)
//   K  -> bf16 [b][p][d]   (row-major: score B-frags contiguous)
//   Vt -> bf16 [b][d][p]   (transposed: context B-frags contiguous)
// grid = (PLEN/16, NPROT) ; block = 256 (8 waves, wave w owns d-tile w)
// TDM path: wave 0 DMAs (A, Wk, Wv) f32 tiles into double-buffered LDS with
// TENSOR_LOAD_TO_LDS, overlapped with WMMA on the other buffer.
// ---------------------------------------------------------------------------
__global__ void kv_kernel(const float* __restrict__ prot,
                          const float* __restrict__ Wk,
                          const float* __restrict__ bk,
                          const float* __restrict__ Wv,
                          const float* __restrict__ bv,
                          unsigned short* __restrict__ Kbf,
                          unsigned short* __restrict__ Vt) {
    const int tid   = threadIdx.x;
    const int dt    = tid >> 5;
    const int lane  = tid & 31;
    const int mlane = lane & 15;
    const int half  = lane >> 4;
    const int p0    = blockIdx.x * 16;
    const int b     = blockIdx.y;

    const float* Arow = prot + ((size_t)b * PLEN + p0) * PDIM;

    v8f kacc = zero8();
    v8f vacc = zero8();

#ifdef HAVE_TDM
    // Double-buffered f32 staging filled by the Tensor Data Mover.
    // Per buffer: A[16][32] (512 f) + K[32][128] (4096 f) + V[32][128] (4096 f)
    __shared__ float smem[2 * (512 + 4096 + 4096)];   // 69,632 bytes
    const unsigned BUF_BYTES = (512 + 4096 + 4096) * 4u;   // 34,816
    const unsigned OFF_A = 0u, OFF_K = 512u * 4u, OFF_V = (512u + 4096u) * 4u;
    const int NSTEP = PDIM / 32;   // 40

    if (dt == 0) {   // wave 0 drives the DMA pipeline
        unsigned base = 0u;   // step 0 -> buffer 0
        tdm_load_2d(base + OFF_A, Arow,       32, 16, (unsigned)PDIM, 16,  (unsigned)PDIM);
        tdm_load_2d(base + OFF_K, Wk,        128, 32, 128, (unsigned)PDIM, 128);
        tdm_load_2d(base + OFF_V, Wv,        128, 32, 128, (unsigned)PDIM, 128);
    }

    for (int s = 0; s < NSTEP; ++s) {
        if (dt == 0) {
            if (s + 1 < NSTEP) {
                const int e1 = (s + 1) * 32;
                unsigned base = (unsigned)(((s + 1) & 1) * BUF_BYTES);
                tdm_load_2d(base + OFF_A, Arow + e1,            32, 16,
                            (unsigned)(PDIM - e1), 16, (unsigned)PDIM);
                tdm_load_2d(base + OFF_K, Wk + (size_t)e1 * LDIM, 128, 32,
                            128, (unsigned)(PDIM - e1), 128);
                tdm_load_2d(base + OFF_V, Wv + (size_t)e1 * LDIM, 128, 32,
                            128, (unsigned)(PDIM - e1), 128);
                tdm_wait3();   // 6 outstanding -> wait for current buffer's 3
            } else {
                tdm_wait0();   // drain tail
            }
        }
        __syncthreads();       // current buffer ready for everyone

        const float* Af = &smem[(s & 1) * (512 + 4096 + 4096)];
        const float* Kf = Af + 512;
        const float* Vf = Af + 512 + 4096;

        v16us a, kb, vb;
        const int dcol = dt * 16 + mlane;
#pragma unroll
        for (int j = 0; j < 16; ++j) {
            int kA = 16 * half + j;
            a[j]  = f2b(Af[mlane * 32 + kA]);
            int kB = 16 * half + j;
            kb[j] = f2b(Kf[kB * 128 + dcol]);
            vb[j] = f2b(Vf[kB * 128 + dcol]);
        }
        kacc = wmma_bf16(a, kb, kacc);
        vacc = wmma_bf16(a, vb, vacc);
        __syncthreads();       // all readers done before buffer is re-filled
    }
#else
    // Fallback: cooperative VALU staging (round-1 path)
    __shared__ unsigned short sA [16 * 32];
    __shared__ unsigned short sKT[128 * 32];
    __shared__ unsigned short sVT[128 * 32];

    for (int e0 = 0; e0 < PDIM; e0 += 32) {
#pragma unroll
        for (int i = 0; i < 2; ++i) {
            int idx = tid + 256 * i;
            int m = idx >> 5, k = idx & 31;
            sA[idx] = f2b(Arow[(size_t)m * PDIM + e0 + k]);
        }
#pragma unroll
        for (int i = 0; i < 16; ++i) {
            int idx = tid + 256 * i;
            int k = idx >> 7, d = idx & 127;
            sKT[d * 32 + k] = f2b(Wk[(size_t)(e0 + k) * LDIM + d]);
            sVT[d * 32 + k] = f2b(Wv[(size_t)(e0 + k) * LDIM + d]);
        }
        __syncthreads();

        v8us  alo = *(const v8us*)&sA[mlane * 32 + 16 * half];
        v8us  ahi = *(const v8us*)&sA[mlane * 32 + 16 * half + 8];
        v16us a   = pack16(alo, ahi);
        v16us kb  = *(const v16us*)&sKT[(dt * 16 + mlane) * 32 + 16 * half];
        v16us vb  = *(const v16us*)&sVT[(dt * 16 + mlane) * 32 + 16 * half];

        kacc = wmma_bf16(a, kb, kacc);
        vacc = wmma_bf16(a, vb, vacc);
        __syncthreads();
    }
#endif

    const int d = dt * 16 + mlane;
    const float bkd = bk[d];
    const float bvd = bv[d];
#pragma unroll
    for (int r = 0; r < 8; ++r) {
        int p = p0 + r + 8 * half;
        Kbf[((size_t)b * PLEN + p) * LDIM + d] = f2b(kacc[r] + bkd);
        Vt [((size_t)b * LDIM + d) * PLEN + p] = f2b(vacc[r] + bvd);
    }
}

// ---------------------------------------------------------------------------
// Kernel 4: flash attention per (graph, 16-node tile) + residual
// grid = (64, NPROT) ; block = 32 (one wave)
// ---------------------------------------------------------------------------
__global__ void attn_kernel(const float* __restrict__ h_lig,
                            const unsigned short* __restrict__ Qbf,
                            const unsigned short* __restrict__ Kbf,
                            const unsigned short* __restrict__ Vt,
                            const int* __restrict__ seg,
                            float* __restrict__ out) {
    const int b    = blockIdx.y;
    const int s    = seg[b];
    const int e    = seg[b + 1];
    const int node0 = s + (int)blockIdx.x * 16;
    if (node0 >= e) return;                 // uniform -> EXEC stays all-ones

    __shared__ unsigned short sP[16 * 32];  // P tile staged for A-frag transpose

    const int lane  = threadIdx.x;
    const int mlane = lane & 15;
    const int half  = lane >> 4;

    // resident Q A-fragments (rows clamped; invalid rows masked at the store)
    int qrow = node0 + mlane;
    if (qrow > N_NODES - 1) qrow = N_NODES - 1;
    v16us qf[4];
#pragma unroll
    for (int kt = 0; kt < 4; ++kt) {
        const unsigned short* qb = &Qbf[(size_t)qrow * LDIM + kt * 32];
        v8us lo = *(const v8us*)&qb[16 * half];
        v8us hi = *(const v8us*)&qb[16 * half + 8];
        qf[kt] = pack16(lo, hi);
    }

    const unsigned short* Kb = Kbf + (size_t)b * PLEN * LDIM;
    const unsigned short* Vb = Vt  + (size_t)b * LDIM * PLEN;
    const float scale = 0.08838834764831845f;  // 1/sqrt(128)

    float mrun[8], lrun[8];
    v8f   ctx[8];
#pragma unroll
    for (int r = 0; r < 8; ++r) { mrun[r] = -1e30f; lrun[r] = 0.0f; ctx[r] = zero8(); }

    for (int p0 = 0; p0 < PLEN; p0 += 32) {
        // ---- scores: S[16 nodes x 32 positions] via 8 WMMAs over d=128 ----
        v8f S0 = zero8();
        v8f S1 = zero8();
#pragma unroll
        for (int kt = 0; kt < 4; ++kt) {
            v16us kb0 = *(const v16us*)&Kb[(size_t)(p0 + mlane)      * LDIM + kt * 32 + 16 * half];
            v16us kb1 = *(const v16us*)&Kb[(size_t)(p0 + 16 + mlane) * LDIM + kt * 32 + 16 * half];
            S0 = wmma_bf16(qf[kt], kb0, S0);
            S1 = wmma_bf16(qf[kt], kb1, S1);
        }

        // ---- online softmax update (row-wise; 16-lane reductions) ----
        float corr[8];
#pragma unroll
        for (int r = 0; r < 8; ++r) {
            float a0 = S0[r] * scale;
            float a1 = S1[r] * scale;
            float mx = fmaxf(a0, a1);
#pragma unroll
            for (int o = 1; o < 16; o <<= 1) mx = fmaxf(mx, __shfl_xor(mx, o, 32));
            float mn  = fmaxf(mrun[r], mx);
            float c   = __expf(mrun[r] - mn);
            float e0v = __expf(a0 - mn);
            float e1v = __expf(a1 - mn);
            float rs  = e0v + e1v;
#pragma unroll
            for (int o = 1; o < 16; o <<= 1) rs += __shfl_xor(rs, o, 32);
            lrun[r] = lrun[r] * c + rs;
            mrun[r] = mn;
            corr[r] = c;
            sP[(r + 8 * half) * 32 + mlane]      = f2b(e0v);
            sP[(r + 8 * half) * 32 + 16 + mlane] = f2b(e1v);
        }
#pragma unroll
        for (int dt = 0; dt < 8; ++dt)
#pragma unroll
            for (int r = 0; r < 8; ++r) ctx[dt][r] *= corr[r];

        __syncthreads();

        v8us  plo = *(const v8us*)&sP[mlane * 32 + 16 * half];
        v8us  phi = *(const v8us*)&sP[mlane * 32 + 16 * half + 8];
        v16us pa  = pack16(plo, phi);

        // ---- context: ctx[16 x 128] += P[16 x 32] @ V[32 x 128] ----
#pragma unroll
        for (int dt = 0; dt < 8; ++dt) {
            v16us vb = *(const v16us*)&Vb[(size_t)(dt * 16 + mlane) * PLEN + p0 + 16 * half];
            ctx[dt] = wmma_bf16(pa, vb, ctx[dt]);
        }
        __syncthreads();
    }

    // ---- epilogue: normalize, residual add, masked store ----
    int nv = e - node0; if (nv > 16) nv = 16;
#pragma unroll
    for (int r = 0; r < 8; ++r) {
        int row = r + 8 * half;
        if (row < nv) {
            float inv = 1.0f / lrun[r];
            int node = node0 + row;
#pragma unroll
            for (int dt = 0; dt < 8; ++dt) {
                int d = dt * 16 + mlane;
                out[(size_t)node * LDIM + d] =
                    h_lig[(size_t)node * LDIM + d] + ctx[dt][r] * inv;
            }
        }
    }
}

// ---------------------------------------------------------------------------
// Host launcher
// ---------------------------------------------------------------------------
extern "C" void kernel_launch(void* const* d_in, const int* in_sizes, int n_in,
                              void* d_out, int out_size, void* d_ws, size_t ws_size,
                              hipStream_t stream) {
    (void)in_sizes; (void)n_in; (void)out_size; (void)ws_size;

    const float* h_lig = (const float*)d_in[0];
    const float* prot  = (const float*)d_in[1];
    const int*   batch = (const int*)  d_in[2];
    const float* Wq    = (const float*)d_in[3];
    const float* bq    = (const float*)d_in[4];
    const float* Wk    = (const float*)d_in[5];
    const float* bk    = (const float*)d_in[6];
    const float* Wv    = (const float*)d_in[7];
    const float* bv    = (const float*)d_in[8];
    float* out = (float*)d_out;

    // workspace layout (bf16 bits stored as ushort)
    const size_t kvBytes = (size_t)NPROT * PLEN * LDIM * sizeof(unsigned short); // 2 MB
    const size_t qBytes  = (size_t)N_NODES * LDIM * sizeof(unsigned short);      // 256 KB
    char* ws = (char*)d_ws;
    unsigned short* Kbf = (unsigned short*)(ws);
    unsigned short* Vt  = (unsigned short*)(ws + kvBytes);
    unsigned short* Qbf = (unsigned short*)(ws + 2 * kvBytes);
    int*            seg = (int*)           (ws + 2 * kvBytes + qBytes);

    seg_kernel<<<1, 32, 0, stream>>>(batch, seg);
    q_kernel<<<N_NODES / 16, 256, 0, stream>>>(h_lig, Wq, bq, Qbf);
    kv_kernel<<<dim3(PLEN / 16, NPROT), 256, 0, stream>>>(prot, Wk, bk, Wv, bv, Kbf, Vt);
    attn_kernel<<<dim3(N_NODES / 16, NPROT), 32, 0, stream>>>(h_lig, Qbf, Kbf, Vt, seg, out);
}
